// SparseDecoderWave_50852412785483
// MI455X (gfx1250) — matrix operand order repaired and checked by reading, not compiled, not run on hardware
//
#include <hip/hip_runtime.h>

// ---------------------------------------------------------------------------
// Types for gfx1250 WMMA (wave32): D(16x16 f32) = A(16x32 f16) x B(32x16 f16) + C
// ---------------------------------------------------------------------------
typedef __attribute__((ext_vector_type(16))) _Float16 v16h;
typedef __attribute__((ext_vector_type(8)))  _Float16 v8h;
typedef __attribute__((ext_vector_type(8)))  float    v8f;

#define PAD_ZERO    0
#define PAD_REFLECT 1
#define PAD_EDGE    2

// Branchless, compile-time-mode padding. Returns clamped/reflected index and
// sets ok=false only for zero-pad out-of-range (address stays valid).
template <int MODE>
__device__ __forceinline__ int pad_idx(int i, int n, bool& ok) {
  if (MODE == PAD_ZERO) {
    ok = ((unsigned)i < (unsigned)n);
    return min(max(i, 0), n - 1);
  } else if (MODE == PAD_REFLECT) {
    ok = true;
    i = (i < 0) ? -i : i;
    return (i >= n) ? (2 * n - 2 - i) : i;
  } else {  // PAD_EDGE
    ok = true;
    return min(max(i, 0), n - 1);
  }
}

// ---------------------------------------------------------------------------
// fp32 -> fp16 convert (activations)
// ---------------------------------------------------------------------------
__global__ void f2h_kernel(const float* __restrict__ x, _Float16* __restrict__ y, long n) {
  long i = (long)blockIdx.x * blockDim.x + threadIdx.x;
  if (i < n) y[i] = (_Float16)x[i];
}

// fp32 OIHW weights -> f16 RSC layout: dst[o][(r*3+s)*Cpad + c], channel-padded
// with zeros to Cpad (multiple of 64). K' = 9*Cpad is a multiple of 64.
__global__ void wconv_rsc_kernel(const float* __restrict__ src, _Float16* __restrict__ dst,
                                 int Cout, int Cin, int Cpad) {
  long idx = (long)blockIdx.x * blockDim.x + threadIdx.x;
  long total = (long)Cout * 9 * Cpad;
  if (idx >= total) return;
  int c = (int)(idx % Cpad);
  long t = idx / Cpad;
  int rs = (int)(t % 9);
  int o  = (int)(t / 9);
  dst[idx] = (c < Cin) ? (_Float16)src[((size_t)o * Cin + c) * 9 + rs] : (_Float16)0.0f;
}

// ---------------------------------------------------------------------------
// Implicit-GEMM 3x3 conv via v_wmma_f32_16x16x32_f16.
//   X  : [Nb][Cin][H][W]      f16 (NCHW)
//   Wt : [Cout][9*Cpad]       f16, RSC layout, channel-padded
//   Y  : [Nb][Cout][H][W]     f16
// Block: 256 thr = 8 waves (4 M x 2 N); tile = 64 oc x 64 px; K chunk = 64.
// Each wave: 16x32 via two accumulators sharing one A fragment (4 WMMA/chunk).
// Double-buffered LDS. RSC layout => per-chunk-uniform (r,s): B gather is a
// coalesced strided run of 16 consecutive channels, ~1 VALU op per element.
// ---------------------------------------------------------------------------
#define TILE_M 64
#define TILE_N 64
#define TILE_K 64
#define KPAD   80   // LDS row stride in halves (160B, 32B-aligned, bank-skewed)

template <int PADMODE>
__launch_bounds__(256)
__global__ void conv3x3_wmma(const _Float16* __restrict__ X,
                             const _Float16* __restrict__ Wt,
                             const float* __restrict__ bias,
                             _Float16* __restrict__ Y,
                             int Nb, int Cin, int Cpad, int H, int W, int Cout,
                             int doLeaky,
                             const float* __restrict__ postMask) {
  const int Kp   = 9 * Cpad;              // padded K extent (multiple of 64)
  const int tid  = threadIdx.x;
  const int wave = tid >> 5;
  const int lane = tid & 31;
  const int waveM = wave >> 1;   // 0..3
  const int waveN = wave & 1;    // 0..1

  const int pix0 = blockIdx.x * TILE_N;   // 64 consecutive pixels (may span rows)
  const int m0   = blockIdx.y * TILE_M;

  __shared__ _Float16 As[2][TILE_M * KPAD];
  __shared__ _Float16 Bs[2][TILE_N * KPAD];

  // ---- A staging coords: row-contiguous copies, coalesced along K ----
  const int arow = tid >> 2;                    // 0..63
  const int aseg = (tid & 3) * 16;              // halves
  const int amc  = min(m0 + arow, Cout - 1);    // clamp: extra rows discarded later

  // ---- B staging coords: lanes sweep x (coalesced), tid>>6 picks c-group ----
  const int bp    = tid & 63;                   // pixel within tile
  const int bkoff = (tid >> 6) * 16;            // 16 halves (channels) per thread
  const int bpix  = pix0 + bp;
  const int bxx   = bpix % W;
  const int bt    = bpix / W;
  const int byy   = bt % H;
  const int bnn   = bt / H;
  const _Float16* Xbn = X + (size_t)bnn * Cin * H * W;
  const size_t chanStep = (size_t)H * W;

  // ---- stage A (weights): unconditional 2x16B copy per thread ----
  auto stageA = [&](int buf, int k0) {
    const uint4* s = (const uint4*)(Wt + (size_t)amc * (size_t)Kp + k0 + aseg);
    __builtin_prefetch((const void*)((const char*)s + TILE_K * 2), 0, 1);
    uint4 t0 = s[0];
    uint4 t1 = s[1];
    uint4* d = (uint4*)(&As[buf][arow * KPAD + aseg]);
    d[0] = t0;
    d[1] = t1;
  };

  // ---- stage B (im2col gather): (r,s) uniform per chunk; 16 channels/thread ----
  auto stageB = [&](int buf, int k0) {
    const int rs = k0 / Cpad;                 // uniform: chunk lies in one segment
    const int c0 = k0 - rs * Cpad + bkoff;    // first channel for this thread
    const int r  = (rs >= 6) ? 2 : ((rs >= 3) ? 1 : 0);
    const int s  = rs - 3 * r;
    bool okY, okX;
    const int yy = pad_idx<PADMODE>(byy + r - 1, H, okY);
    const int xs = pad_idx<PADMODE>(bxx + s - 1, W, okX);
    const bool okP = okY && okX;
    const int cc0 = min(c0, Cin - 1);         // keep base address valid
    const _Float16* src = Xbn + ((size_t)cc0 * H + yy) * W + xs;
    v8h bv0, bv1;
#pragma unroll
    for (int i = 0; i < 16; ++i) {
      bool ok = okP && (c0 + i < Cin);
      _Float16 val = src[(ok ? (size_t)i : (size_t)0) * chanStep];
      _Float16 res = ok ? val : (_Float16)0.0f;
      if (i < 8) bv0[i] = res; else bv1[i - 8] = res;
    }
    *(v8h*)(&Bs[buf][bp * KPAD + bkoff])     = bv0;
    *(v8h*)(&Bs[buf][bp * KPAD + bkoff + 8]) = bv1;
  };

  // ---- fragment coordinates (A/B: same contiguous k = 16*(lane/16) + s order; C: ISA 16x16 layout) ----
  const int mrow  = waveM * 16 + (lane & 15);
  const int prow0 = waveN * 32 + (lane & 15);
  const int prow1 = prow0 + 16;
  const int abase = (lane >> 4) * 16;
  const int bbase = (lane >> 4) * 16;

  v8f acc[2] = {};
  const int nChunks = Kp / TILE_K;

  stageA(0, 0);
  stageB(0, 0);
  __syncthreads();

  for (int i = 0; i < nChunks; ++i) {
    const int p = i & 1;
    if (i + 1 < nChunks) {          // stage next chunk into the other buffer
      stageA(1 - p, (i + 1) * TILE_K);
      stageB(1 - p, (i + 1) * TILE_K);
    }
#pragma unroll
    for (int kc = 0; kc < TILE_K; kc += 32) {
      const _Float16* ap = &As[p][mrow * KPAD + kc];
      v8h alo = *(const v8h*)(ap + abase);
      v8h ahi = *(const v8h*)(ap + 8 + abase);
      v16h a;
#pragma unroll
      for (int j = 0; j < 8; ++j) { a[j] = alo[j]; a[j + 8] = ahi[j]; }
      v16h b0 = *(const v16h*)(&Bs[p][prow0 * KPAD + kc + bbase]);
      v16h b1 = *(const v16h*)(&Bs[p][prow1 * KPAD + kc + bbase]);
      acc[0] = __builtin_amdgcn_wmma_f32_16x16x32_f16(
                   false, a, false, b0, (short)0, acc[0], false, false);
      acc[1] = __builtin_amdgcn_wmma_f32_16x16x32_f16(
                   false, a, false, b1, (short)0, acc[1], false, false);
    }
    __syncthreads();
  }

  // ---- epilogue: bias + leaky + spatial mask, f16 store ----
#pragma unroll
  for (int f = 0; f < 2; ++f) {
    int pcol = waveN * 32 + (lane & 15) + f * 16;
    int pidx = pix0 + pcol;
    int xo = pidx % W;
    int t2 = pidx / W;
    int yo = t2 % H;
    int no = t2 / H;
    float mk = postMask ? postMask[((size_t)no * H + yo) * W + xo] : 1.0f;
#pragma unroll
    for (int r = 0; r < 8; ++r) {
      int m  = waveM * 16 + r + ((lane >> 4) * 8);
      int oc = m0 + m;
      if (oc < Cout) {
        float v = acc[f][r] + bias[oc];
        if (doLeaky) v = (v > 0.0f) ? v : 0.2f * v;
        v *= mk;
        Y[(((size_t)no * Cout + oc) * H + yo) * W + xo] = (_Float16)v;
      }
    }
  }
}

// ---------------------------------------------------------------------------
// Small conv heads (Cout=1 or 3): scalar, fused scale + optional half-res mask
// ---------------------------------------------------------------------------
template <int PADMODE>
__global__ void conv3x3_small(const _Float16* __restrict__ X,
                              const float* __restrict__ Wt,  // [O][Cin][3][3] fp32
                              const float* __restrict__ bias,
                              float* __restrict__ Y,
                              int Nb, int Cin, int H, int W, int O,
                              float scale,
                              const float* __restrict__ halfMask) {
  int idx = blockIdx.x * blockDim.x + threadIdx.x;
  int total = Nb * O * H * W;
  if (idx >= total) return;
  int x = idx % W; int t = idx / W;
  int y = t % H;   t /= H;
  int o = t % O;   int n = t / O;

  // precompute padded coords + validity once (9 taps)
  int ys[3], xs[3];
  float wy[3], wx[3];
#pragma unroll
  for (int r = 0; r < 3; ++r) {
    bool ok;
    ys[r] = pad_idx<PADMODE>(y + r - 1, H, ok);
    wy[r] = ok ? 1.0f : 0.0f;
    xs[r] = pad_idx<PADMODE>(x + r - 1, W, ok);
    wx[r] = ok ? 1.0f : 0.0f;
  }

  const _Float16* Xn = X + (size_t)n * Cin * H * W;
  const float* wb = Wt + (size_t)o * Cin * 9;
  float acc = 0.0f;
  for (int c = 0; c < Cin; ++c) {
    const _Float16* xc = Xn + (size_t)c * H * W;
    const float* wc = wb + c * 9;
#pragma unroll
    for (int r = 0; r < 3; ++r) {
#pragma unroll
      for (int s = 0; s < 3; ++s) {
        acc += (float)xc[(size_t)ys[r] * W + xs[s]] * wc[r * 3 + s] * (wy[r] * wx[s]);
      }
    }
  }
  float v = (acc + bias[o]) * scale;
  if (halfMask)
    v *= halfMask[((size_t)n * (H / 2) + (y >> 1)) * (W / 2) + (x >> 1)];
  Y[idx] = v;
}

// ---------------------------------------------------------------------------
// Inverse Haar DWT: ll [N,H,W] + h [N,3,H,W] -> out [N,2H,2W]
// ---------------------------------------------------------------------------
__global__ void idwt_haar_kernel(const float* __restrict__ ll,
                                 const float* __restrict__ h,
                                 float* __restrict__ out,
                                 int Nb, int H, int W) {
  int idx = blockIdx.x * blockDim.x + threadIdx.x;
  int total = Nb * H * W;
  if (idx >= total) return;
  int x = idx % W; int t = idx / W;
  int y = t % H;   int n = t / H;
  size_t hw = (size_t)H * W;
  float LL = ll[(size_t)n * hw + (size_t)y * W + x];
  const float* hn = h + (size_t)n * 3 * hw;
  float LH = hn[(size_t)y * W + x];
  float HL = hn[hw + (size_t)y * W + x];
  float HH = hn[2 * hw + (size_t)y * W + x];
  float x00 = (LL + LH + HL + HH) * 0.5f;
  float x01 = (LL - LH + HL - HH) * 0.5f;
  float x10 = (LL + LH - HL - HH) * 0.5f;
  float x11 = (LL - LH - HL + HH) * 0.5f;
  float* on = out + (size_t)n * 4 * hw;
  int W2 = 2 * W;
  on[(size_t)(2 * y) * W2 + 2 * x]         = x00;
  on[(size_t)(2 * y) * W2 + 2 * x + 1]     = x01;
  on[(size_t)(2 * y + 1) * W2 + 2 * x]     = x10;
  on[(size_t)(2 * y + 1) * W2 + 2 * x + 1] = x11;
}

// ---------------------------------------------------------------------------
// Global min/max (order-preserving uint encoding) + threshold mask
// ---------------------------------------------------------------------------
__device__ __forceinline__ unsigned enc_f(float f) {
  unsigned u = __float_as_uint(f);
  return (u & 0x80000000u) ? ~u : (u | 0x80000000u);
}
__device__ __forceinline__ float dec_f(unsigned u) {
  unsigned b = (u & 0x80000000u) ? (u & 0x7FFFFFFFu) : ~u;
  return __uint_as_float(b);
}

__global__ void init_minmax_kernel(unsigned* mm) {
  if (threadIdx.x == 0) { mm[0] = 0u; mm[1] = 0xFFFFFFFFu; }
}

__global__ void reduce_minmax_kernel(const float* __restrict__ x, int n, unsigned* mm) {
  __shared__ unsigned smax[256];
  __shared__ unsigned smin[256];
  unsigned lmax = 0u, lmin = 0xFFFFFFFFu;
  for (int i = blockIdx.x * blockDim.x + threadIdx.x; i < n; i += gridDim.x * blockDim.x) {
    unsigned e = enc_f(x[i]);
    lmax = max(lmax, e);
    lmin = min(lmin, e);
  }
  smax[threadIdx.x] = lmax; smin[threadIdx.x] = lmin;
  __syncthreads();
  for (int s = 128; s > 0; s >>= 1) {
    if (threadIdx.x < (unsigned)s) {
      smax[threadIdx.x] = max(smax[threadIdx.x], smax[threadIdx.x + s]);
      smin[threadIdx.x] = min(smin[threadIdx.x], smin[threadIdx.x + s]);
    }
    __syncthreads();
  }
  if (threadIdx.x == 0) { atomicMax(&mm[0], smax[0]); atomicMin(&mm[1], smin[0]); }
}

__global__ void make_mask_kernel(const float* __restrict__ h, int Nb, int H, int W,
                                 const unsigned* __restrict__ mm, float tr,
                                 float* __restrict__ mask) {
  int idx = blockIdx.x * blockDim.x + threadIdx.x;
  int total = Nb * H * W;
  if (idx >= total) return;
  int x = idx % W; int t = idx / W;
  int y = t % H;   int n = t / H;
  float thresh = (dec_f(mm[0]) - dec_f(mm[1])) * tr;
  size_t hw = (size_t)H * W;
  const float* hn = h + (size_t)n * 3 * hw;
  float a = fabsf(hn[(size_t)y * W + x]);
  float b = fabsf(hn[hw + (size_t)y * W + x]);
  float c = fabsf(hn[2 * hw + (size_t)y * W + x]);
  mask[idx] = (fmaxf(a, fmaxf(b, c)) > thresh) ? 1.0f : 0.0f;
}

// OR-maxpool kxk (SAME) at mask resolution
__global__ void pool_or_kernel(const float* __restrict__ m, int Nb, int H, int W,
                               int k, float* __restrict__ out) {
  int idx = blockIdx.x * blockDim.x + threadIdx.x;
  int total = Nb * H * W;
  if (idx >= total) return;
  int x = idx % W; int t = idx / W;
  int y = t % H;   int n = t / H;
  int r = k >> 1;
  const float* mn = m + (size_t)n * H * W;
  float v = 0.0f;
  for (int dy = -r; dy <= r; ++dy) {
    int yy = y + dy; if (yy < 0 || yy >= H) continue;
    for (int dx = -r; dx <= r; ++dx) {
      int xx = x + dx; if (xx < 0 || xx >= W) continue;
      if (mn[(size_t)yy * W + xx] > 0.0f) v = 1.0f;
    }
  }
  out[idx] = v;
}

// OR-maxpool kxk (SAME) over up2x(mask): out at [2H,2W]
__global__ void pool_or_up_kernel(const float* __restrict__ m, int Nb, int H, int W,
                                  int k, float* __restrict__ out) {
  int H2 = 2 * H, W2 = 2 * W;
  int idx = blockIdx.x * blockDim.x + threadIdx.x;
  int total = Nb * H2 * W2;
  if (idx >= total) return;
  int x = idx % W2; int t = idx / W2;
  int y = t % H2;   int n = t / H2;
  int r = k >> 1;
  const float* mn = m + (size_t)n * H * W;
  float v = 0.0f;
  for (int dy = -r; dy <= r; ++dy) {
    int yy = y + dy; if (yy < 0 || yy >= H2) continue;
    for (int dx = -r; dx <= r; ++dx) {
      int xx = x + dx; if (xx < 0 || xx >= W2) continue;
      if (mn[(size_t)(yy >> 1) * W + (xx >> 1)] > 0.0f) v = 1.0f;
    }
  }
  out[idx] = v;
}

// ---------------------------------------------------------------------------
// Feature preparation (fused concat / nearest-up2x / sparse masks), f16 output
// ---------------------------------------------------------------------------
__global__ void prep_feat1_kernel(const _Float16* __restrict__ xd0,  // [N,C0,H/2,W/2]
                                  const float* __restrict__ xskip,   // [N,C1,H,W]
                                  _Float16* __restrict__ feat,       // [N,C0+C1,H,W]
                                  int Nb, int C0, int C1, int H, int W) {
  long idx = (long)blockIdx.x * blockDim.x + threadIdx.x;
  long total = (long)Nb * (C0 + C1) * H * W;
  if (idx >= total) return;
  int x = (int)(idx % W); long t = idx / W;
  int y = (int)(t % H);   t /= H;
  int c = (int)(t % (C0 + C1));
  int n = (int)(t / (C0 + C1));
  float v;
  if (c < C0)
    v = (float)xd0[(((size_t)n * C0 + c) * (H / 2) + (y >> 1)) * (W / 2) + (x >> 1)];
  else
    v = xskip[(((size_t)n * C1 + (c - C0)) * H + y) * W + x];
  feat[idx] = (_Float16)v;
}

__global__ void prep_feat_masked_kernel(const _Float16* __restrict__ xlow, // [N,C0,H/2,W/2]
                                        const float* __restrict__ xskip,  // [N,C1,H,W]
                                        const float* __restrict__ up_m,   // [N,H/2,W/2]
                                        const float* __restrict__ conva_m,// [N,H,W]
                                        _Float16* __restrict__ feat,      // [N,C0+C1,H,W]
                                        int Nb, int C0, int C1, int H, int W) {
  long idx = (long)blockIdx.x * blockDim.x + threadIdx.x;
  long total = (long)Nb * (C0 + C1) * H * W;
  if (idx >= total) return;
  int x = (int)(idx % W); long t = idx / W;
  int y = (int)(t % H);   t /= H;
  int c = (int)(t % (C0 + C1));
  int n = (int)(t / (C0 + C1));
  float v;
  if (c < C0) {
    size_t hoff = ((size_t)n * (H / 2) + (y >> 1)) * (W / 2) + (x >> 1);
    v = (float)xlow[(((size_t)n * C0 + c) * (H / 2) + (y >> 1)) * (W / 2) + (x >> 1)] *
        up_m[hoff];
  } else {
    v = xskip[(((size_t)n * C1 + (c - C0)) * H + y) * W + x];
  }
  v *= conva_m[((size_t)n * H + y) * W + x];
  feat[idx] = (_Float16)v;
}

// ---------------------------------------------------------------------------
// Host orchestration
// ---------------------------------------------------------------------------
static inline int cdiv(long a, long b) { return (int)((a + b - 1) / b); }
static inline int cpad64(int c) { return (c + 63) & ~63; }

extern "C" void kernel_launch(void* const* d_in, const int* in_sizes, int n_in,
                              void* d_out, int out_size, void* d_ws, size_t ws_size,
                              hipStream_t stream) {
  (void)in_sizes; (void)n_in; (void)out_size; (void)ws_size;

  const float* x32    = (const float*)d_in[0];
  const float* x16    = (const float*)d_in[1];
  const float* x8     = (const float*)d_in[2];
  const float* x4     = (const float*)d_in[3];
  const float* conv2w = (const float*)d_in[4];  const float* conv2b = (const float*)d_in[5];
  const float* up1w   = (const float*)d_in[6];  const float* up1b   = (const float*)d_in[7];
  const float* w1llw  = (const float*)d_in[8];  const float* w1llb  = (const float*)d_in[9];
  const float* w1w    = (const float*)d_in[10]; const float* w1b    = (const float*)d_in[11];
  const float* up2w   = (const float*)d_in[12]; const float* up2b   = (const float*)d_in[13];
  const float* w2w    = (const float*)d_in[14]; const float* w2b    = (const float*)d_in[15];
  const float* up3w   = (const float*)d_in[16]; const float* up3b   = (const float*)d_in[17];
  const float* w3w    = (const float*)d_in[18]; const float* w3b    = (const float*)d_in[19];

  constexpr int Nb = 4;
  constexpr int H1 = 10,  W1 = 32;    // x32 / x_d0
  constexpr int H2 = 20,  W2 = 64;    // x_d1 level
  constexpr int H3 = 40,  W3 = 128;   // xa level
  constexpr int H4 = 80,  W4 = 256;   // xb level
  constexpr int C32 = 2208, C16 = 384, C8 = 192, C4 = 96;
  constexpr int F = 1104;             // conv2 out; up1 = 552; up2 = 276; up3 = 138
  constexpr int Ci1 = C32;            // conv inputs
  constexpr int Ci2 = F + C16;        // 1488
  constexpr int Ci3 = F / 2 + C8;     // 744
  constexpr int Ci4 = F / 4 + C4;     // 372
  const int Cp1 = cpad64(Ci1), Cp2 = cpad64(Ci2), Cp3 = cpad64(Ci3), Cp4 = cpad64(Ci4);

  // ---- carve workspace ----
  char* ws = (char*)d_ws;
  size_t off = 0;
  auto alloc = [&](size_t bytes) -> char* {
    off = (off + 255) & ~size_t(255);
    char* p = ws + off;
    off += bytes;
    return p;
  };
  _Float16* wq1    = (_Float16*)alloc((size_t)F * 9 * Cp1 * 2);
  _Float16* wq2    = (_Float16*)alloc((size_t)(F / 2) * 9 * Cp2 * 2);
  _Float16* wq3    = (_Float16*)alloc((size_t)(F / 4) * 9 * Cp3 * 2);
  _Float16* wq4    = (_Float16*)alloc((size_t)(F / 8) * 9 * Cp4 * 2);
  _Float16* x32h   = (_Float16*)alloc((size_t)Nb * C32 * H1 * W1 * 2);
  _Float16* xd0h   = (_Float16*)alloc((size_t)Nb * F * H1 * W1 * 2);
  _Float16* feat1h = (_Float16*)alloc((size_t)Nb * Ci2 * H2 * W2 * 2);
  _Float16* xd1h   = (_Float16*)alloc((size_t)Nb * (F / 2) * H2 * W2 * 2);
  float*    ll1    = (float*)alloc((size_t)Nb * H2 * W2 * 4);
  float*    h1     = (float*)alloc((size_t)Nb * 3 * H2 * W2 * 4);
  float*    ll2    = (float*)alloc((size_t)Nb * H3 * W3 * 4);
  float*    mask1  = (float*)alloc((size_t)Nb * H2 * W2 * 4);
  float*    upm1   = (float*)alloc((size_t)Nb * H2 * W2 * 4);
  float*    conva1 = (float*)alloc((size_t)Nb * H3 * W3 * 4);
  float*    wavem1 = (float*)alloc((size_t)Nb * H3 * W3 * 4);
  _Float16* feat2h = (_Float16*)alloc((size_t)Nb * Ci3 * H3 * W3 * 2);
  _Float16* xah    = (_Float16*)alloc((size_t)Nb * (F / 4) * H3 * W3 * 2);
  float*    h2     = (float*)alloc((size_t)Nb * 3 * H3 * W3 * 4);
  float*    ll3    = (float*)alloc((size_t)Nb * H4 * W4 * 4);
  float*    mask2  = (float*)alloc((size_t)Nb * H3 * W3 * 4);
  float*    upm2   = (float*)alloc((size_t)Nb * H3 * W3 * 4);
  float*    conva2 = (float*)alloc((size_t)Nb * H4 * W4 * 4);
  float*    wavem2 = (float*)alloc((size_t)Nb * H4 * W4 * 4);
  _Float16* feat3h = (_Float16*)alloc((size_t)Nb * Ci4 * H4 * W4 * 2);
  _Float16* xbh    = (_Float16*)alloc((size_t)Nb * (F / 8) * H4 * W4 * 2);
  float*    h3     = (float*)alloc((size_t)Nb * 3 * H4 * W4 * 4);
  unsigned* mm     = (unsigned*)alloc(2 * sizeof(unsigned));

  const int TB = 256;

  // ---- 0) f16 conversions (weights -> RSC, channel-padded) ----
  wconv_rsc_kernel<<<cdiv((long)F * 9 * Cp1, TB), TB, 0, stream>>>(conv2w, wq1, F, Ci1, Cp1);
  wconv_rsc_kernel<<<cdiv((long)(F / 2) * 9 * Cp2, TB), TB, 0, stream>>>(up1w, wq2, F / 2, Ci2, Cp2);
  wconv_rsc_kernel<<<cdiv((long)(F / 4) * 9 * Cp3, TB), TB, 0, stream>>>(up2w, wq3, F / 4, Ci3, Cp3);
  wconv_rsc_kernel<<<cdiv((long)(F / 8) * 9 * Cp4, TB), TB, 0, stream>>>(up3w, wq4, F / 8, Ci4, Cp4);
  f2h_kernel<<<cdiv((long)Nb * C32 * H1 * W1, TB), TB, 0, stream>>>(x32, x32h, (long)Nb * C32 * H1 * W1);

  // ---- 1) x_d0 = conv3x3(x32, conv2_w, edge) ----
  {
    dim3 g((Nb * H1 * W1) / TILE_N, cdiv(F, TILE_M));
    conv3x3_wmma<PAD_EDGE><<<g, TB, 0, stream>>>(x32h, wq1, conv2b, xd0h,
                                                 Nb, Ci1, Cp1, H1, W1, F, 0, nullptr);
  }
  // ---- 2) feat1 = concat(up2x(x_d0), x16); x_d1 = leaky(conv(feat1, reflect)) ----
  prep_feat1_kernel<<<cdiv((long)Nb * Ci2 * H2 * W2, TB), TB, 0, stream>>>(
      xd0h, x16, feat1h, Nb, F, C16, H2, W2);
  {
    dim3 g((Nb * H2 * W2) / TILE_N, cdiv(F / 2, TILE_M));
    conv3x3_wmma<PAD_REFLECT><<<g, TB, 0, stream>>>(feat1h, wq2, up1b, xd1h,
                                                    Nb, Ci2, Cp2, H2, W2, F / 2, 1, nullptr);
  }
  // ---- 3) wavelet heads level 1 + iDWT ----
  conv3x3_small<PAD_EDGE><<<cdiv((long)Nb * 1 * H2 * W2, TB), TB, 0, stream>>>(
      xd1h, w1llw, w1llb, ll1, Nb, F / 2, H2, W2, 1, 8.0f, nullptr);
  conv3x3_small<PAD_ZERO><<<cdiv((long)Nb * 3 * H2 * W2, TB), TB, 0, stream>>>(
      xd1h, w1w, w1b, h1, Nb, F / 2, H2, W2, 3, 4.0f, nullptr);
  idwt_haar_kernel<<<cdiv((long)Nb * H2 * W2, TB), TB, 0, stream>>>(ll1, h1, ll2, Nb, H2, W2);

  // ---- 4) sparse level 1 masks ----
  init_minmax_kernel<<<1, 64, 0, stream>>>(mm);
  reduce_minmax_kernel<<<256, TB, 0, stream>>>(ll2, Nb * H3 * W3, mm);
  make_mask_kernel<<<cdiv((long)Nb * H2 * W2, TB), TB, 0, stream>>>(h1, Nb, H2, W2, mm, 0.1f, mask1);
  pool_or_kernel<<<cdiv((long)Nb * H2 * W2, TB), TB, 0, stream>>>(mask1, Nb, H2, W2, 5, upm1);
  pool_or_up_kernel<<<cdiv((long)Nb * H3 * W3, TB), TB, 0, stream>>>(mask1, Nb, H2, W2, 5, conva1);
  pool_or_up_kernel<<<cdiv((long)Nb * H3 * W3, TB), TB, 0, stream>>>(mask1, Nb, H2, W2, 3, wavem1);

  // ---- 5) xa level ----
  prep_feat_masked_kernel<<<cdiv((long)Nb * Ci3 * H3 * W3, TB), TB, 0, stream>>>(
      xd1h, x8, upm1, conva1, feat2h, Nb, F / 2, C8, H3, W3);
  {
    dim3 g((Nb * H3 * W3) / TILE_N, cdiv(F / 4, TILE_M));
    conv3x3_wmma<PAD_REFLECT><<<g, TB, 0, stream>>>(feat2h, wq3, up2b, xah,
                                                    Nb, Ci3, Cp3, H3, W3, F / 4, 1, wavem1);
  }
  conv3x3_small<PAD_ZERO><<<cdiv((long)Nb * 3 * H3 * W3, TB), TB, 0, stream>>>(
      xah, w2w, w2b, h2, Nb, F / 4, H3, W3, 3, 2.0f, mask1 /*wavelet = up2x(mask1)*/);
  idwt_haar_kernel<<<cdiv((long)Nb * H3 * W3, TB), TB, 0, stream>>>(ll2, h2, ll3, Nb, H3, W3);

  // ---- 6) sparse level 0 masks ----
  init_minmax_kernel<<<1, 64, 0, stream>>>(mm);
  reduce_minmax_kernel<<<256, TB, 0, stream>>>(ll3, Nb * H4 * W4, mm);
  make_mask_kernel<<<cdiv((long)Nb * H3 * W3, TB), TB, 0, stream>>>(h2, Nb, H3, W3, mm, 0.1f, mask2);
  pool_or_kernel<<<cdiv((long)Nb * H3 * W3, TB), TB, 0, stream>>>(mask2, Nb, H3, W3, 5, upm2);
  pool_or_up_kernel<<<cdiv((long)Nb * H4 * W4, TB), TB, 0, stream>>>(mask2, Nb, H3, W3, 5, conva2);
  pool_or_up_kernel<<<cdiv((long)Nb * H4 * W4, TB), TB, 0, stream>>>(mask2, Nb, H3, W3, 3, wavem2);

  // ---- 7) xb level + final iDWT into d_out ----
  prep_feat_masked_kernel<<<cdiv((long)Nb * Ci4 * H4 * W4, TB), TB, 0, stream>>>(
      xah, x4, upm2, conva2, feat3h, Nb, F / 4, C4, H4, W4);
  {
    dim3 g((Nb * H4 * W4) / TILE_N, cdiv(F / 8, TILE_M));
    conv3x3_wmma<PAD_REFLECT><<<g, TB, 0, stream>>>(feat3h, wq4, up3b, xbh,
                                                    Nb, Ci4, Cp4, H4, W4, F / 8, 1, wavem2);
  }
  conv3x3_small<PAD_ZERO><<<cdiv((long)Nb * 3 * H4 * W4, TB), TB, 0, stream>>>(
      xbh, w3w, w3b, h3, Nb, F / 8, H4, W4, 3, 1.0f, mask2);
  idwt_haar_kernel<<<cdiv((long)Nb * H4 * W4, TB), TB, 0, stream>>>(
      ll3, h3, (float*)d_out, Nb, H4, W4);
}
